// SelfAttention_16432544874604
// MI455X (gfx1250) — matrix operand order repaired and checked by reading, not compiled, not run on hardware
//
#include <hip/hip_runtime.h>

#define EMB   1024
#define HEADS 16
#define HD    64
#define BB    4
#define TT    2048
#define MROWS (BB*TT)

typedef __attribute__((ext_vector_type(16))) _Float16 v16h;
typedef __attribute__((ext_vector_type(4)))  _Float16 v4h;
typedef __attribute__((ext_vector_type(8)))  float    v8f;
typedef __attribute__((ext_vector_type(4)))  int      v4i;

union Frag { v16h v; unsigned int u[8]; _Float16 h[16]; };

__device__ __forceinline__ v8f wmma_f16(const Frag& a, const Frag& b, v8f c) {
  // D = A(16x32 f16) * B(32x16 f16) + C(16x16 f32)
  return __builtin_amdgcn_wmma_f32_16x16x32_f16(false, a.v, false, b.v,
                                                (short)0, c, false, false);
}

// ---- CDNA5 async global->LDS copy (ASYNCcnt path), with safe fallback ------
#if defined(__has_builtin)
#if __has_builtin(__builtin_amdgcn_global_load_async_to_lds_b128)
#define USE_ASYNC_LDS 1
#endif
#endif

#ifdef USE_ASYNC_LDS
typedef __attribute__((address_space(1))) v4i* gas_v4i;
typedef __attribute__((address_space(3))) v4i* las_v4i;
#endif

__device__ __forceinline__ void cp16(_Float16* lds, const _Float16* glb) {
#ifdef USE_ASYNC_LDS
  __builtin_amdgcn_global_load_async_to_lds_b128(
      (gas_v4i)(glb), (las_v4i)(lds), 0, 0);
#else
  *(uint4*)lds = *(const uint4*)glb;
#endif
}
__device__ __forceinline__ void wait_cp() {
#ifdef USE_ASYNC_LDS
  asm volatile("s_wait_asynccnt 0x0" ::: "memory");
#endif
}

// ---------------------------------------------------------------- fp32 -> f16
__global__ __launch_bounds__(256) void cvt_f32_f16(const float* __restrict__ in,
                                                   _Float16* __restrict__ out,
                                                   int n) {
  int i = (blockIdx.x * 256 + threadIdx.x) * 4;
  if (i < n) {
    float4 f = *(const float4*)(in + i);
    v4h o = { (_Float16)f.x, (_Float16)f.y, (_Float16)f.z, (_Float16)f.w };
    *(v4h*)(out + i) = o;
  }
}

// ---------------------- weight prep: fp32 [K][N] -> f16 transposed [N][K] ----
__global__ __launch_bounds__(256) void wt_prep(
    const float* __restrict__ Wq, const float* __restrict__ Wk,
    const float* __restrict__ Wv, const float* __restrict__ Wo,
    _Float16* __restrict__ Wt) {
  const int z = blockIdx.y;
  const float* src = (z == 0) ? Wq : (z == 1) ? Wk : (z == 2) ? Wv : Wo;
  _Float16* dst = Wt + (size_t)z * EMB * EMB;
  int g = blockIdx.x * 256 + threadIdx.x;        // 131072 threads total
  int n = g >> 7;                                 // 128 threads per column
  int kc = (g & 127) * 8;
  #pragma unroll
  for (int j = 0; j < 8; ++j)
    dst[(size_t)n * EMB + kc + j] = (_Float16)src[(size_t)(kc + j) * EMB + n];
}

// ------------------------------------------------------- tiled WMMA GEMM
// C[M,N=1024] = A_f16[M,K=1024] @ Wt_f16[N,K]^T + bias
// mode 0/1: f16 [B,H,T,hd] (Q/K). mode 2: f16 V transposed [B,H,hd,T].
// mode 3: f32 row-major to d_out.
__global__ __launch_bounds__(256) void gemm_wmma(
    const _Float16* __restrict__ A, const _Float16* __restrict__ Wt,
    const float* __restrict__ b0, const float* __restrict__ b1,
    const float* __restrict__ b2,
    _Float16* __restrict__ q_out, _Float16* __restrict__ k_out,
    _Float16* __restrict__ v_out,
    float* __restrict__ out_f32, int mode_base)
{
  const int K = EMB, N = EMB;
  const int z = blockIdx.z;
  const int mode = mode_base + z;
  const _Float16* W = Wt + (size_t)z * EMB * EMB;
  const float* bias = (z == 0) ? b0 : (z == 1) ? b1 : b2;
  _Float16* outh = (mode == 0) ? q_out : (mode == 1) ? k_out : v_out;

  __shared__ __align__(16) _Float16 Alds[2][128 * 32];   // [m][k]   8KB x2
  __shared__ __align__(16) _Float16 Blds[2][64 * 32];    // [n][k]   4KB x2

  const int tid  = threadIdx.x;
  const int wave = tid >> 5, lane = tid & 31;
  const int l16  = lane & 15, hi = lane >> 4;
  const int m0 = blockIdx.y * 128, n0 = blockIdx.x * 64;

  // tile-copy lambda: A 128x32 (2 chunks/thread), B 64x32 (1 chunk/thread)
  auto copy_tile = [&](int buf, int k0) {
    #pragma unroll
    for (int i = 0; i < 2; ++i) {
      int c = tid + i * 256;
      int row = c >> 2, cq = c & 3;
      cp16(&Alds[buf][row * 32 + cq * 8],
           A + (size_t)(m0 + row) * K + k0 + cq * 8);
    }
    {
      int n = tid >> 2, cq = tid & 3;
      cp16(&Blds[buf][n * 32 + cq * 8],
           W + (size_t)(n0 + n) * K + k0 + cq * 8);
    }
  };

  v8f acc[4] = {};

  copy_tile(0, 0);
  wait_cp();
  __syncthreads();

  int cur = 0;
  for (int k0 = 0; k0 < K; k0 += 32) {
    if (k0 + 32 < K) copy_tile(cur ^ 1, k0 + 32);   // overlap with WMMA

    Frag af;
    const int mrow = wave * 16 + l16;
    #pragma unroll
    for (int p = 0; p < 8; ++p) {
      int kk = ((p & 3) * 2) + hi * 8 + ((p >> 2) * 16);       // A-layout
      af.u[p] = *(const unsigned int*)&Alds[cur][mrow * 32 + kk];
    }
    #pragma unroll
    for (int j = 0; j < 4; ++j) {
      Frag bf;
      const int ncol = j * 16 + l16;
      #pragma unroll
      for (int p = 0; p < 8; ++p) {
        int kk = ((p & 3) * 2) + hi * 8 + ((p >> 2) * 16);       // B-layout
        bf.u[p] = *(const unsigned int*)&Blds[cur][ncol * 32 + kk];
      }
      acc[j] = wmma_f16(af, bf, acc[j]);
    }
    wait_cp();
    __syncthreads();
    cur ^= 1;
  }

  #pragma unroll
  for (int j = 0; j < 4; ++j) {
    const int n_g = n0 + j * 16 + l16;
    const float bvv = bias[n_g];
    #pragma unroll
    for (int r = 0; r < 8; ++r) {
      const int m_g = m0 + wave * 16 + r + 8 * hi;
      const float val = acc[j][r] + bvv;
      const int b = m_g >> 11, t = m_g & (TT - 1);
      const int h = n_g >> 6,  d = n_g & (HD - 1);
      if (mode < 2) {          // Q,K: [B,H,T,hd]
        outh[((size_t)(b * HEADS + h) * TT + t) * HD + d] = (_Float16)val;
      } else if (mode == 2) {  // V: transposed [B,H,hd,T]
        outh[((size_t)(b * HEADS + h) * HD + d) * TT + t] = (_Float16)val;
      } else {                 // final projection: f32 [B,T,E]
        out_f32[(size_t)m_g * N + n_g] = val;
      }
    }
  }
}

// ------------------------------------------------------- flash attention
// Q/K f16 [B*H, T, HD]; V f16 transposed [B*H, HD, T]; out f16 [B, T, E].
__global__ __launch_bounds__(128) void attn_wmma(
    const _Float16* __restrict__ Qm, const _Float16* __restrict__ Km,
    const _Float16* __restrict__ Vt, _Float16* __restrict__ Oh)
{
  __shared__ __align__(16) _Float16 Plds[4 * 16 * 32];   // per-wave P staging

  const int tid  = threadIdx.x;
  const int wave = tid >> 5, lane = tid & 31;
  const int l16  = lane & 15, hi = lane >> 4;
  const int bh = blockIdx.y;
  const int q0 = blockIdx.x * 64 + wave * 16;
  const size_t base = (size_t)bh * TT * HD;
  _Float16* Pw = Plds + wave * 512;

  // Q fragments (two 32-wide K-chunks over hd=64)
  Frag qf[2];
  #pragma unroll
  for (int c = 0; c < 2; ++c)
    #pragma unroll
    for (int p = 0; p < 8; ++p) {
      int d = c * 32 + ((p & 3) * 2) + hi * 8 + ((p >> 2) * 16);
      qf[c].u[p] = *(const unsigned int*)(Qm + base + (size_t)(q0 + l16) * HD + d);
    }

  v8f o[4] = {};
  float mst[8], lst[8];
  #pragma unroll
  for (int r = 0; r < 8; ++r) { mst[r] = -1e30f; lst[r] = 0.f; }

  const float scale = 0.125f;                 // 1/sqrt(64)
  const int nk = q0 + 16;                     // causal key bound (exclusive)

  for (int kt = 0; kt < nk; kt += 32) {
    if (kt + 32 < nk) {                       // prefetch next K/V tiles
      __builtin_prefetch(Km + base + (size_t)(kt + 32 + lane) * HD, 0, 0);
      __builtin_prefetch(Vt + base + (size_t)lane * TT + kt + 32, 0, 0);
      __builtin_prefetch(Vt + base + (size_t)(lane + 32) * TT + kt + 32, 0, 0);
    }
    // ---- S = Q @ K^T for two 16-key sub-tiles
    v8f s[2] = {};
    #pragma unroll
    for (int j = 0; j < 2; ++j) {
      const int kb = kt + j * 16;
      if (kb < nk) {
        #pragma unroll
        for (int c = 0; c < 2; ++c) {
          Frag kf;                            // K^T as B-matrix: d pairs contiguous
          #pragma unroll
          for (int p = 0; p < 8; ++p) {
            int d = c * 32 + ((p & 3) * 2) + hi * 8 + ((p >> 2) * 16);
            kf.u[p] = *(const unsigned int*)(Km + base + (size_t)(kb + l16) * HD + d);
          }
          s[j] = wmma_f16(qf[c], kf, s[j]);
        }
      }
    }

    // ---- online softmax (row = r + 8*hi, key-col = lane&15 within half)
    #pragma unroll
    for (int r = 0; r < 8; ++r) {
      const int rowq = q0 + r + 8 * hi;
      const int key0 = kt + l16, key1 = kt + 16 + l16;
      float v0 = (key0 <= rowq) ? s[0][r] * scale : -1e30f;
      float v1 = (key1 <= rowq) ? s[1][r] * scale : -1e30f;
      float mx = fmaxf(v0, v1);
      #pragma unroll
      for (int w = 1; w <= 8; w <<= 1) mx = fmaxf(mx, __shfl_xor(mx, w, 32));
      const float mnew  = fmaxf(mst[r], mx);
      const float alpha = __expf(mst[r] - mnew);
      const float p0 = __expf(v0 - mnew);
      const float p1 = __expf(v1 - mnew);
      float ps = p0 + p1;
      #pragma unroll
      for (int w = 1; w <= 8; w <<= 1) ps += __shfl_xor(ps, w, 32);
      lst[r] = lst[r] * alpha + ps;
      mst[r] = mnew;
      #pragma unroll
      for (int j = 0; j < 4; ++j) o[j][r] *= alpha;
      const int prow = (r + 8 * hi) * 32;
      Pw[prow + l16]      = (_Float16)p0;
      Pw[prow + 16 + l16] = (_Float16)p1;
    }
    // same-wave LDS write -> cross-lane read: wait on DS counter only
    asm volatile("s_wait_dscnt 0x0" ::: "memory");

    // ---- P as A-fragment (16q x 32k)
    Frag pf;
    #pragma unroll
    for (int p = 0; p < 8; ++p) {
      int kk = ((p & 3) * 2) + hi * 8 + ((p >> 2) * 16);
      pf.u[p] = *(const unsigned int*)&Pw[l16 * 32 + kk];
    }
    // ---- O += P @ V  (V transposed: key pairs contiguous -> b32 loads)
    #pragma unroll
    for (int j = 0; j < 4; ++j) {
      Frag vf;
      const int d = j * 16 + l16;
      #pragma unroll
      for (int p = 0; p < 8; ++p) {
        int kl = ((p & 3) * 2) + hi * 8 + ((p >> 2) * 16);
        vf.u[p] = *(const unsigned int*)(Vt + base + (size_t)d * TT + kt + kl);
      }
      o[j] = wmma_f16(pf, vf, o[j]);
    }
  }

  // ---- epilogue: O /= l, store f16 [B,T,E]
  const int b = bh >> 4, h = bh & 15;
  float rl[8];
  #pragma unroll
  for (int r = 0; r < 8; ++r) rl[r] = 1.f / lst[r];
  #pragma unroll
  for (int j = 0; j < 4; ++j)
    #pragma unroll
    for (int r = 0; r < 8; ++r) {
      const int row = q0 + r + 8 * hi;
      const int d = j * 16 + l16;
      Oh[((size_t)(b * TT + row)) * EMB + h * HD + d] = (_Float16)(o[j][r] * rl[r]);
    }
}

// ----------------------------------------------------------------- launcher
extern "C" void kernel_launch(void* const* d_in, const int* in_sizes, int n_in,
                              void* d_out, int out_size, void* d_ws, size_t ws_size,
                              hipStream_t stream) {
  const float* x  = (const float*)d_in[0];
  const float* Wq = (const float*)d_in[1];
  const float* bq = (const float*)d_in[2];
  const float* Wk = (const float*)d_in[3];
  const float* bk = (const float*)d_in[4];
  const float* Wv = (const float*)d_in[5];
  const float* bv = (const float*)d_in[6];
  const float* Wo = (const float*)d_in[7];
  const float* bo = (const float*)d_in[8];
  float* out = (float*)d_out;

  char* ws = (char*)d_ws;
  const size_t SZ = (size_t)MROWS * EMB * sizeof(_Float16);   // 16 MB each
  _Float16* xh = (_Float16*)(ws);
  _Float16* qh = (_Float16*)(ws + SZ);
  _Float16* kh = (_Float16*)(ws + 2 * SZ);
  _Float16* vt = (_Float16*)(ws + 3 * SZ);
  _Float16* ah = (_Float16*)(ws + 4 * SZ);
  _Float16* wt = (_Float16*)(ws + 5 * SZ);                    // 4 x 2 MB

  const int n = MROWS * EMB;
  cvt_f32_f16<<<n / 1024, 256, 0, stream>>>(x, xh, n);
  wt_prep<<<dim3(512, 4), 256, 0, stream>>>(Wq, Wk, Wv, Wo, wt);

  gemm_wmma<<<dim3(EMB / 64, MROWS / 128, 3), 256, 0, stream>>>(
      xh, wt, bq, bk, bv, qh, kh, vt, nullptr, 0);

  attn_wmma<<<dim3(TT / 64, BB * HEADS), 128, 0, stream>>>(qh, kh, vt, ah);

  gemm_wmma<<<dim3(EMB / 64, MROWS / 128, 1), 256, 0, stream>>>(
      ah, wt + 3 * (size_t)EMB * EMB, bo, bo, bo,
      nullptr, nullptr, nullptr, out, 3);
}
